// QuestionAnsweringBase_4587025072286
// MI455X (gfx1250) — compile-verified, matrix operand order checked by reading, not run on hardware
//
#include <hip/hip_runtime.h>
#include <hip/hip_bf16.h>
#include <math.h>

typedef __attribute__((ext_vector_type(16))) _Float16 v16h;
typedef __attribute__((ext_vector_type(8)))  _Float16 v8h;
typedef __attribute__((ext_vector_type(8)))  float    v8f;

constexpr int DMODEL = 768;
constexpr int FFDIM  = 3072;
constexpr int SEQ    = 1024;
constexpr int BATCH  = 8;
constexpr int NHEAD  = 12;
constexpr int DH     = 64;
constexpr int MROWS  = BATCH * SEQ;   // 8192
constexpr int NBLK   = 64;            // S / 16
constexpr int BSZ    = 16;
constexpr int NMID   = 62;
constexpr int KB     = 7;             // 3 window + 2 global + 2 random

__device__ __forceinline__ void async_b128(unsigned lds_off, const void* gaddr)
{
    asm volatile("global_load_async_to_lds_b128 %0, %1, off"
                 :: "v"(lds_off), "v"(gaddr) : "memory");
}
__device__ __forceinline__ void wait_async0()
{
    asm volatile("s_wait_asynccnt 0x0" ::: "memory");
}

// ---------------------------------------------------------------------------
// WMMA GEMM:  out = act(A[f16, MxK] @ Bt[f16, stored NxK] + bias), f32 and/or
// f16 outputs.  Block tile 128x128 (8 waves, 4x2; each wave 32x64 = 2x4 WMMA
// tiles), BK = 32.  Double-buffered LDS staging via
// global_load_async_to_lds_b128 (ASYNCcnt) overlapped with the WMMAs.
// ---------------------------------------------------------------------------
constexpr int BM = 128, BN = 128, BK = 32;
constexpr int LDA_S = 40;  // padded half-stride (80B, 16B aligned, conflict-free)
constexpr int LDB_S = 40;

__global__ __launch_bounds__(256)
void gemm_f16_wmma(const _Float16* __restrict__ A, const _Float16* __restrict__ Bt,
                   const float* __restrict__ bias,
                   float* __restrict__ out32, _Float16* __restrict__ out16,
                   int M, int N, int K, int act)
{
    __shared__ _Float16 sA[2][BM * LDA_S];
    __shared__ _Float16 sB[2][BN * LDB_S];

    const int tid  = threadIdx.x;
    const int lane = tid & 31;
    const int wid  = tid >> 5;
    const int wm   = wid & 3;        // wave row (0..3) -> 32 output rows each
    const int wn   = wid >> 2;       // wave col (0..1) -> 64 output cols each
    const int bm   = blockIdx.y * BM;
    const int bn   = blockIdx.x * BN;

    const int khalf = (lane >> 4) * 8;   // K-half select per 16-bit WMMA layout
    const int mr    = lane & 15;

    // per-thread staging coordinates: 512 16B chunks per 128x32-half tile
    const int srow0 = tid >> 2;            // chunk rows for t = 0
    const int srow1 = (tid + 256) >> 2;    // chunk rows for t = 1
    const int sseg  = tid & 3;

    const unsigned ldsA0 = (unsigned)(uintptr_t)(&sA[0][srow0 * LDA_S + sseg * 8]);
    const unsigned ldsA1 = (unsigned)(uintptr_t)(&sA[0][srow1 * LDA_S + sseg * 8]);
    const unsigned ldsB0 = (unsigned)(uintptr_t)(&sB[0][srow0 * LDB_S + sseg * 8]);
    const unsigned ldsB1 = (unsigned)(uintptr_t)(&sB[0][srow1 * LDB_S + sseg * 8]);
    const unsigned bufAstep = (unsigned)(BM * LDA_S * sizeof(_Float16));
    const unsigned bufBstep = (unsigned)(BN * LDB_S * sizeof(_Float16));

    const _Float16* gA0 = A  + (size_t)(bm + srow0) * K + sseg * 8;
    const _Float16* gA1 = A  + (size_t)(bm + srow1) * K + sseg * 8;
    const _Float16* gB0 = Bt + (size_t)(bn + srow0) * K + sseg * 8;
    const _Float16* gB1 = Bt + (size_t)(bn + srow1) * K + sseg * 8;

    auto stage = [&](int buf, int k0) {
        const unsigned aoff = buf * bufAstep;
        const unsigned boff = buf * bufBstep;
        async_b128(ldsA0 + aoff, gA0 + k0);
        async_b128(ldsA1 + aoff, gA1 + k0);
        async_b128(ldsB0 + boff, gB0 + k0);
        async_b128(ldsB1 + boff, gB1 + k0);
    };

    v8f acc[2][4];
    #pragma unroll
    for (int i = 0; i < 2; ++i)
        #pragma unroll
        for (int j = 0; j < 4; ++j)
            acc[i][j] = v8f{};

    const int T = K / BK;
    stage(0, 0);

    for (int kt = 0; kt < T; ++kt) {
        const int cur = kt & 1;
        wait_async0();        // this wave's async writes to buf[cur] done
        __syncthreads();      // everyone's writes visible; prior reads finished
        if (kt + 1 < T) stage(1 - cur, (kt + 1) * BK);   // prefetch next tile

        const _Float16* bAp = &sA[cur][0];
        const _Float16* bBp = &sB[cur][0];

        v16h afrag[2], bfrag[4];
        #pragma unroll
        for (int i = 0; i < 2; ++i) {
            const _Float16* p = bAp + (wm * 32 + i * 16 + mr) * LDA_S;
            v8h lo = *(const v8h*)(p + khalf);
            v8h hi = *(const v8h*)(p + khalf + 16);
            afrag[i] = __builtin_shufflevector(lo, hi, 0,1,2,3,4,5,6,7,8,9,10,11,12,13,14,15);
        }
        #pragma unroll
        for (int j = 0; j < 4; ++j) {
            const _Float16* p = bBp + (wn * 64 + j * 16 + mr) * LDB_S;
            v8h lo = *(const v8h*)(p + khalf);
            v8h hi = *(const v8h*)(p + khalf + 16);
            bfrag[j] = __builtin_shufflevector(lo, hi, 0,1,2,3,4,5,6,7,8,9,10,11,12,13,14,15);
        }
        #pragma unroll
        for (int i = 0; i < 2; ++i)
            #pragma unroll
            for (int j = 0; j < 4; ++j)
                acc[i][j] = __builtin_amdgcn_wmma_f32_16x16x32_f16(
                    false, afrag[i], false, bfrag[j], (short)0, acc[i][j], false, false);
    }

    // epilogue: C layout -> lane&15 = N, VGPR r = M (+8 for lanes 16..31)
    const int hi16 = lane >> 4;
    #pragma unroll
    for (int i = 0; i < 2; ++i) {
        #pragma unroll
        for (int j = 0; j < 4; ++j) {
            const int n  = bn + wn * 64 + j * 16 + mr;
            const float bv = bias ? bias[n] : 0.0f;
            #pragma unroll
            for (int r = 0; r < 8; ++r) {
                const int m = bm + wm * 32 + i * 16 + r + 8 * hi16;
                float v = acc[i][j][r] + bv;
                if (act == 1) {   // tanh-approx GELU (jax.nn.gelu default)
                    float u = 0.7978845608028654f * (v + 0.044715f * v * v * v);
                    v = 0.5f * v * (1.0f + tanhf(u));
                }
                if (out32) out32[(size_t)m * N + n] = v;
                if (out16) out16[(size_t)m * N + n] = (_Float16)v;
            }
        }
    }
}

// ---------------------------------------------------------------------------
// Middle-block sparse attention: one workgroup per (b, h, mid block)
// ---------------------------------------------------------------------------
__global__ __launch_bounds__(256)
void attn_mid(const _Float16* __restrict__ q16, const _Float16* __restrict__ k16,
              const _Float16* __restrict__ v16, const int* __restrict__ rb,
              _Float16* __restrict__ attn16)
{
    const int mi  = blockIdx.x;      // 0..61
    const int h   = blockIdx.y;
    const int b   = blockIdx.z;
    const int mid = mi + 1;

    int idx[KB];
    idx[0] = mid - 1; idx[1] = mid; idx[2] = mid + 1;
    idx[3] = 0;       idx[4] = NBLK - 1;
    idx[5] = rb[mi * 2 + 0]; idx[6] = rb[mi * 2 + 1];
    bool dup[KB];
    #pragma unroll
    for (int j = 0; j < KB; ++j) {
        bool d = false;
        for (int jj = 0; jj < j; ++jj) d = d || (idx[jj] == idx[j]);
        dup[j] = d;
    }

    __shared__ float    qs[BSZ][DH];
    __shared__ _Float16 ks[KB * BSZ][DH];
    __shared__ _Float16 vs[KB * BSZ][DH];
    __shared__ float    sc[BSZ][KB * BSZ];

    const int tid = threadIdx.x;
    // gathered K/V blocks: contiguous 128B rows -> async DMA straight to LDS
    for (int e = tid; e < KB * BSZ * 8; e += 256) {    // 896 x 16B chunks
        int kk = e >> 3, seg = e & 7;
        int j = kk >> 4, r = kk & 15;
        size_t src = ((size_t)b * SEQ + idx[j] * BSZ + r) * DMODEL + h * DH + seg * 8;
        async_b128((unsigned)(uintptr_t)(&ks[kk][seg * 8]), k16 + src);
        async_b128((unsigned)(uintptr_t)(&vs[kk][seg * 8]), v16 + src);
    }
    for (int e = tid; e < BSZ * DH; e += 256) {
        int r = e >> 6, d = e & 63;
        qs[r][d] = (float)q16[((size_t)b * SEQ + mid * BSZ + r) * DMODEL + h * DH + d];
    }
    wait_async0();
    __syncthreads();

    for (int e = tid; e < BSZ * KB * BSZ; e += 256) {
        int m = e / 112, kc = e % 112;
        float s = 0.0f;
        #pragma unroll
        for (int d = 0; d < DH; ++d) s += qs[m][d] * (float)ks[kc][d];
        s *= 0.125f;
        if (dup[kc >> 4]) s = -1.0e9f;
        sc[m][kc] = s;
    }
    __syncthreads();

    // softmax: wave handles 2 rows, 16 lanes per row
    const int lane = tid & 31;
    const int row  = (tid >> 5) * 2 + (lane >> 4);
    const int c    = lane & 15;
    float vreg[7];
    float mx = -1.0e30f;
    #pragma unroll
    for (int j = 0; j < 7; ++j) { vreg[j] = sc[row][c + 16 * j]; mx = fmaxf(mx, vreg[j]); }
    #pragma unroll
    for (int msk = 1; msk <= 8; msk <<= 1) mx = fmaxf(mx, __shfl_xor(mx, msk, 32));
    float sum = 0.0f;
    #pragma unroll
    for (int j = 0; j < 7; ++j) { vreg[j] = expf(vreg[j] - mx); sum += vreg[j]; }
    #pragma unroll
    for (int msk = 1; msk <= 8; msk <<= 1) sum += __shfl_xor(sum, msk, 32);
    const float inv = 1.0f / sum;
    #pragma unroll
    for (int j = 0; j < 7; ++j) sc[row][c + 16 * j] = vreg[j] * inv;
    __syncthreads();

    for (int e = tid; e < BSZ * DH; e += 256) {
        int m = e >> 6, d = e & 63;
        float acc = 0.0f;
        #pragma unroll 8
        for (int k = 0; k < KB * BSZ; ++k) acc += sc[m][k] * (float)vs[k][d];
        attn16[((size_t)b * SEQ + mid * BSZ + m) * DMODEL + h * DH + d] = (_Float16)acc;
    }
}

// ---------------------------------------------------------------------------
// Global blocks (0 and 63) attend to all 1024 keys; one WG per (b, h, g)
// ---------------------------------------------------------------------------
__global__ __launch_bounds__(256)
void attn_global(const _Float16* __restrict__ q16, const _Float16* __restrict__ k16,
                 const _Float16* __restrict__ v16, _Float16* __restrict__ attn16)
{
    const int g = blockIdx.x;            // 0 -> block 0, 1 -> block 63
    const int h = blockIdx.y;
    const int b = blockIdx.z;
    const int qbase = g ? (SEQ - BSZ) : 0;

    __shared__ float    qs[BSZ][DH];        //  4 KB
    __shared__ _Float16 chunk[128][DH];     // 16 KB
    __shared__ _Float16 p16[BSZ][SEQ];      // 32 KB

    const int tid = threadIdx.x;
    for (int e = tid; e < BSZ * DH; e += 256) {
        int r = e >> 6, d = e & 63;
        qs[r][d] = (float)q16[((size_t)b * SEQ + qbase + r) * DMODEL + h * DH + d];
    }
    __syncthreads();

    const int m = tid >> 4;      // query row (16 threads per row)
    const int c = tid & 15;
    float sc[64];

    #pragma unroll
    for (int ch = 0; ch < 8; ++ch) {
        #pragma unroll
        for (int t = 0; t < 4; ++t) {
            int c4  = tid + t * 256;
            int row = c4 >> 3;
            int seg = c4 & 7;
            async_b128((unsigned)(uintptr_t)(&chunk[row][seg * 8]),
                       k16 + ((size_t)b * SEQ + ch * 128 + row) * DMODEL + h * DH + seg * 8);
        }
        wait_async0();
        __syncthreads();
        #pragma unroll
        for (int jl = 0; jl < 8; ++jl) {
            int kl = c + 16 * jl;
            float s = 0.0f;
            #pragma unroll
            for (int d = 0; d < DH; ++d) s += qs[m][d] * (float)chunk[kl][d];
            sc[ch * 8 + jl] = s * 0.125f;
        }
        __syncthreads();
    }

    const int lane = tid & 31;
    float mx = -1.0e30f;
    #pragma unroll
    for (int t = 0; t < 64; ++t) mx = fmaxf(mx, sc[t]);
    #pragma unroll
    for (int msk = 1; msk <= 8; msk <<= 1) mx = fmaxf(mx, __shfl_xor(mx, msk, 32));
    float sum = 0.0f;
    #pragma unroll
    for (int t = 0; t < 64; ++t) { sc[t] = expf(sc[t] - mx); sum += sc[t]; }
    #pragma unroll
    for (int msk = 1; msk <= 8; msk <<= 1) sum += __shfl_xor(sum, msk, 32);
    const float inv = 1.0f / sum;
    #pragma unroll
    for (int ch = 0; ch < 8; ++ch)
        #pragma unroll
        for (int jl = 0; jl < 8; ++jl)
            p16[m][ch * 128 + c + 16 * jl] = (_Float16)(sc[ch * 8 + jl] * inv);
    __syncthreads();

    // out = P @ V ; thread handles fixed d, rows mb + {0,4,8,12}
    const int d  = tid & 63;
    const int mb = tid >> 6;
    float oacc[4] = {0.f, 0.f, 0.f, 0.f};
    for (int ch = 0; ch < 8; ++ch) {
        #pragma unroll
        for (int t = 0; t < 4; ++t) {
            int c4  = tid + t * 256;
            int row = c4 >> 3;
            int seg = c4 & 7;
            async_b128((unsigned)(uintptr_t)(&chunk[row][seg * 8]),
                       v16 + ((size_t)b * SEQ + ch * 128 + row) * DMODEL + h * DH + seg * 8);
        }
        wait_async0();
        __syncthreads();
        for (int kl = 0; kl < 128; ++kl) {
            float vv = (float)chunk[kl][d];
            #pragma unroll
            for (int i = 0; i < 4; ++i)
                oacc[i] += (float)p16[mb + i * 4][ch * 128 + kl] * vv;
        }
        __syncthreads();
    }
    #pragma unroll
    for (int i = 0; i < 4; ++i)
        attn16[((size_t)b * SEQ + qbase + mb + i * 4) * DMODEL + h * DH + d] = (_Float16)oacc[i];
}

// ---------------------------------------------------------------------------
// residual add + layernorm, dual f32/f16 output.  One block per row.
// ---------------------------------------------------------------------------
__global__ __launch_bounds__(256)
void add_ln(const float* __restrict__ hin, const float* __restrict__ t,
            const float* __restrict__ g, const float* __restrict__ bta,
            float* __restrict__ hout, _Float16* __restrict__ x16)
{
    const int r   = blockIdx.x;
    const int tid = threadIdx.x;
    __shared__ float red[256];

    float v[3];
    #pragma unroll
    for (int i = 0; i < 3; ++i) {
        int cidx = tid + i * 256;
        v[i] = hin[(size_t)r * DMODEL + cidx] + t[(size_t)r * DMODEL + cidx];
    }
    float ls = v[0] + v[1] + v[2];
    red[tid] = ls; __syncthreads();
    for (int s = 128; s > 0; s >>= 1) { if (tid < s) red[tid] += red[tid + s]; __syncthreads(); }
    const float mean = red[0] * (1.0f / DMODEL);
    __syncthreads();

    float lv = 0.0f;
    #pragma unroll
    for (int i = 0; i < 3; ++i) { float d = v[i] - mean; lv += d * d; }
    red[tid] = lv; __syncthreads();
    for (int s = 128; s > 0; s >>= 1) { if (tid < s) red[tid] += red[tid + s]; __syncthreads(); }
    const float rs = rsqrtf(red[0] * (1.0f / DMODEL) + 1e-12f);

    #pragma unroll
    for (int i = 0; i < 3; ++i) {
        int cidx = tid + i * 256;
        float y = (v[i] - mean) * rs * g[cidx] + bta[cidx];
        hout[(size_t)r * DMODEL + cidx] = y;
        x16 [(size_t)r * DMODEL + cidx] = (_Float16)y;
    }
}

__global__ void init_h(const float* __restrict__ x, float* __restrict__ h32,
                       _Float16* __restrict__ x16, size_t n)
{
    size_t i = (size_t)blockIdx.x * 256 + threadIdx.x;
    if (i < n) { float v = x[i]; h32[i] = v; x16[i] = (_Float16)v; }
}

// convert W[K][N] f32 -> Wt[N][K] f16 (transposed for WMMA B fragments)
__global__ void cvt_w(const float* __restrict__ W, _Float16* __restrict__ Wt, int K, int N)
{
    size_t e = (size_t)blockIdx.x * 256 + threadIdx.x;
    if (e >= (size_t)K * N) return;
    int n = (int)(e / K), k = (int)(e % K);
    Wt[e] = (_Float16)W[(size_t)k * N + n];
}

__global__ __launch_bounds__(256)
void logits_kernel(const float* __restrict__ h, const float* __restrict__ wsv,
                   const float* __restrict__ wev, float* __restrict__ out)
{
    const int row  = blockIdx.x * 8 + (threadIdx.x >> 5);
    const int lane = threadIdx.x & 31;
    float s = 0.0f, e = 0.0f;
    for (int c = lane; c < DMODEL; c += 32) {
        float v = h[(size_t)row * DMODEL + c];
        s += v * wsv[c];
        e += v * wev[c];
    }
    #pragma unroll
    for (int msk = 16; msk >= 1; msk >>= 1) {
        s += __shfl_xor(s, msk, 32);
        e += __shfl_xor(e, msk, 32);
    }
    if (lane == 0) { out[row] = s; out[MROWS + row] = e; }
}

// ---------------------------------------------------------------------------
extern "C" void kernel_launch(void* const* d_in, const int* in_sizes, int n_in,
                              void* d_out, int out_size, void* d_ws, size_t ws_size,
                              hipStream_t stream)
{
    const float* x   = (const float*)d_in[0];
    const int*   rb  = (const int*)d_in[1];
    const float* Wq  = (const float*)d_in[2];
    const float* bq  = (const float*)d_in[3];
    const float* Wk  = (const float*)d_in[4];
    const float* bk  = (const float*)d_in[5];
    const float* Wv  = (const float*)d_in[6];
    const float* bv  = (const float*)d_in[7];
    const float* Wo  = (const float*)d_in[8];
    const float* bo  = (const float*)d_in[9];
    const float* l1g = (const float*)d_in[10];
    const float* l1b = (const float*)d_in[11];
    const float* W1  = (const float*)d_in[12];
    const float* b1  = (const float*)d_in[13];
    const float* W2  = (const float*)d_in[14];
    const float* b2  = (const float*)d_in[15];
    const float* l2g = (const float*)d_in[16];
    const float* l2b = (const float*)d_in[17];
    const float* wst = (const float*)d_in[18];
    const float* wen = (const float*)d_in[19];

    char*  ws  = (char*)d_ws;
    size_t off = 0;
    auto take = [&](size_t bytes) -> void* {
        void* p = ws + off;
        off += (bytes + 255) & ~(size_t)255;
        return p;
    };
    float*    h32    = (float*)   take((size_t)MROWS * DMODEL * 4);
    float*    tmp32  = (float*)   take((size_t)MROWS * DMODEL * 4);
    _Float16* x16    = (_Float16*)take((size_t)MROWS * DMODEL * 2);
    _Float16* q16    = (_Float16*)take((size_t)MROWS * DMODEL * 2);
    _Float16* k16    = (_Float16*)take((size_t)MROWS * DMODEL * 2);
    _Float16* v16    = (_Float16*)take((size_t)MROWS * DMODEL * 2);
    _Float16* attn16 = (_Float16*)take((size_t)MROWS * DMODEL * 2);
    _Float16* ff16   = (_Float16*)take((size_t)MROWS * FFDIM  * 2);
    _Float16* wt16   = (_Float16*)take((size_t)DMODEL * FFDIM * 2);

    const size_t nel = (size_t)MROWS * DMODEL;
    init_h<<<(int)((nel + 255) / 256), 256, 0, stream>>>(x, h32, x16, nel);

    const dim3 gemmDD(DMODEL / BN, MROWS / BM);   // N=768
    const dim3 gemmDF(FFDIM  / BN, MROWS / BM);   // N=3072
    const int  cvtDD = (DMODEL * DMODEL + 255) / 256;
    const int  cvtDF = (DMODEL * FFDIM  + 255) / 256;

    for (int l = 0; l < 2; ++l) {
        const size_t wdd = (size_t)l * DMODEL * DMODEL;
        const size_t wdf = (size_t)l * DMODEL * FFDIM;
        const size_t vd  = (size_t)l * DMODEL;
        const size_t vf  = (size_t)l * FFDIM;

        // Q, K, V projections (f16 out)
        cvt_w<<<cvtDD, 256, 0, stream>>>(Wq + wdd, wt16, DMODEL, DMODEL);
        gemm_f16_wmma<<<gemmDD, 256, 0, stream>>>(x16, wt16, bq + vd, nullptr, q16,
                                                  MROWS, DMODEL, DMODEL, 0);
        cvt_w<<<cvtDD, 256, 0, stream>>>(Wk + wdd, wt16, DMODEL, DMODEL);
        gemm_f16_wmma<<<gemmDD, 256, 0, stream>>>(x16, wt16, bk + vd, nullptr, k16,
                                                  MROWS, DMODEL, DMODEL, 0);
        cvt_w<<<cvtDD, 256, 0, stream>>>(Wv + wdd, wt16, DMODEL, DMODEL);
        gemm_f16_wmma<<<gemmDD, 256, 0, stream>>>(x16, wt16, bv + vd, nullptr, v16,
                                                  MROWS, DMODEL, DMODEL, 0);

        // block-sparse attention
        attn_mid   <<<dim3(NMID, NHEAD, BATCH), 256, 0, stream>>>(q16, k16, v16, rb, attn16);
        attn_global<<<dim3(2,    NHEAD, BATCH), 256, 0, stream>>>(q16, k16, v16, attn16);

        // output projection + residual + LN1
        cvt_w<<<cvtDD, 256, 0, stream>>>(Wo + wdd, wt16, DMODEL, DMODEL);
        gemm_f16_wmma<<<gemmDD, 256, 0, stream>>>(attn16, wt16, bo + vd, tmp32, nullptr,
                                                  MROWS, DMODEL, DMODEL, 0);
        add_ln<<<MROWS, 256, 0, stream>>>(h32, tmp32, l1g + vd, l1b + vd, h32, x16);

        // FFN
        cvt_w<<<cvtDF, 256, 0, stream>>>(W1 + wdf, wt16, DMODEL, FFDIM);
        gemm_f16_wmma<<<gemmDF, 256, 0, stream>>>(x16, wt16, b1 + vf, nullptr, ff16,
                                                  MROWS, FFDIM, DMODEL, 1);
        cvt_w<<<cvtDF, 256, 0, stream>>>(W2 + wdf, wt16, FFDIM, DMODEL);
        gemm_f16_wmma<<<gemmDD, 256, 0, stream>>>(ff16, wt16, b2 + vd, tmp32, nullptr,
                                                  MROWS, DMODEL, FFDIM, 0);
        add_ln<<<MROWS, 256, 0, stream>>>(h32, tmp32, l2g + vd, l2b + vd, h32, x16);
    }

    logits_kernel<<<MROWS / 8, 256, 0, stream>>>(h32, wst, wen, (float*)d_out);
}
